// ClassificationTransformer_60808146977066
// MI455X (gfx1250) — matrix operand order repaired and checked by reading, not compiled, not run on hardware
//
#include <hip/hip_runtime.h>

// ---------------------------------------------------------------------------
// Types
// ---------------------------------------------------------------------------
typedef __bf16 bf16_t;
typedef __attribute__((ext_vector_type(16))) __bf16 v16bf;
typedef __attribute__((ext_vector_type(8)))  float  v8f;

union Frag16 { v16bf v; unsigned u[8]; };

__device__ __forceinline__ bf16_t f2bf(float f) {
    unsigned x = __builtin_bit_cast(unsigned, f);
    unsigned r = (x + 0x7FFFu + ((x >> 16) & 1u)) >> 16;
    unsigned short h = (unsigned short)r;
    return __builtin_bit_cast(bf16_t, h);
}
__device__ __forceinline__ float bf2f(bf16_t b) {
    unsigned short h = __builtin_bit_cast(unsigned short, b);
    unsigned x = ((unsigned)h) << 16;
    return __builtin_bit_cast(float, x);
}

// ---------------------------------------------------------------------------
// Async global->LDS copy path (CDNA5 GLOBAL_LOAD_ASYNC_TO_LDS_B128, ASYNCcnt)
// with synchronous fallback when the builtins are unavailable.
// Builtin parameter types (per hipcc diagnostic): v4i addrspace(1)* src,
// v4i addrspace(3)* dst, imm offset, imm cpol.
// ---------------------------------------------------------------------------
#if defined(__has_builtin)
#  if __has_builtin(__builtin_amdgcn_global_load_async_to_lds_b128) && \
      __has_builtin(__builtin_amdgcn_s_wait_asynccnt)
#    define USE_ASYNC_LDS 1
#  endif
#endif
#ifndef USE_ASYNC_LDS
#  define USE_ASYNC_LDS 0
#endif

#if USE_ASYNC_LDS
typedef int v4i __attribute__((vector_size(16)));
typedef __attribute__((address_space(1))) v4i* glb_v4i_t;
typedef __attribute__((address_space(3))) v4i* lds_v4i_t;
__device__ __forceinline__ lds_v4i_t lds_cast(void* p) {
    // generic->LDS addrspace cast == low 32 bits (LDS aperture truncation)
    return (lds_v4i_t)(unsigned)(uintptr_t)p;
}
__device__ __forceinline__ glb_v4i_t glb_cast(const void* p) {
    // generic->global addrspace cast is the identity
    return (glb_v4i_t)(uintptr_t)p;
}
#define ASYNC_WAIT(n) __builtin_amdgcn_s_wait_asynccnt(n)
#else
#define ASYNC_WAIT(n) ((void)0)
#endif

__device__ __forceinline__ void copy16_async(void* lds, const void* glb) {
#if USE_ASYNC_LDS
    __builtin_amdgcn_global_load_async_to_lds_b128(glb_cast(glb), lds_cast(lds), 0, 0);
#else
    *(uint4*)lds = *(const uint4*)glb;
#endif
}

// ---------------------------------------------------------------------------
// Model constants
// ---------------------------------------------------------------------------
#define TT   43
#define HH   1024
#define DKK  512
#define DVV  512
#define FFF  4096
#define NB   512
#define MROW (NB * TT)   /* 22016 = 128 * 172 */

// ---------------------------------------------------------------------------
// Weight prep: f32 (K,N) row-major -> bf16 (N,K) row-major (transposed).
// Makes both GEMM operands contiguous along K for 16B async staging.
// ---------------------------------------------------------------------------
__global__ void __launch_bounds__(256)
cvt_transpose_kernel(const float* __restrict__ src, bf16_t* __restrict__ dst,
                     int K, int N) {
    const int total = K * N;
    for (int i = blockIdx.x * 256 + threadIdx.x; i < total;
         i += gridDim.x * 256) {
        const int n = i / K;
        const int k = i - n * K;
        dst[i] = f2bf(src[(size_t)k * N + n]);
    }
}

// ---------------------------------------------------------------------------
// Embedding: x[row][h] = emb[tok[row]][h] + pos[row % T][h]  (f32 + bf16)
// ---------------------------------------------------------------------------
__global__ void __launch_bounds__(256)
embed_kernel(const int* __restrict__ tok, const float* __restrict__ emb,
             const float* __restrict__ pos, float* __restrict__ xf,
             bf16_t* __restrict__ xb) {
    const int row = blockIdx.x;
    const int t   = row % TT;
    const int tk  = tok[row];
    const float* e = emb + (size_t)tk * HH;
    const float* p = pos + (size_t)t  * HH;
    for (int i = threadIdx.x; i < HH; i += 256) {
        float v = e[i] + p[i];
        xf[(size_t)row * HH + i] = v;
        xb[(size_t)row * HH + i] = f2bf(v);
    }
}

// ---------------------------------------------------------------------------
// WMMA GEMM: C(M,N) = epi(A(M,K) @ Bt(N,K)^T + bias[, + resid])
//   bf16 A (row-major) and Bt (weights pre-transposed, N-major), f32 accum.
//   Block = 256 threads (8 waves). Tile 128x128, k-step 64, double-buffered
//   LDS staged with GLOBAL_LOAD_ASYNC_TO_LDS_B128. Wave -> 32x64 quadrant
//   (2x4 of 16x16x32 WMMA tiles). M,N multiples of 128; K multiple of 64.
// EPI: 0 = bias            -> store bf16
//      1 = bias + residual -> store f32
//      2 = bias + ReLU     -> store bf16
// ---------------------------------------------------------------------------
template <int EPI>
__global__ void __launch_bounds__(256)
gemm_wmma_kernel(const bf16_t* __restrict__ A, const bf16_t* __restrict__ Bt,
                 const float* __restrict__ bias, const float* __restrict__ resid,
                 float* __restrict__ Cf, bf16_t* __restrict__ Cb,
                 int M, int N, int K) {
    __shared__ bf16_t As[2][128 * 64];   // [buf][row][k]
    __shared__ bf16_t Bs[2][128 * 64];   // [buf][col][k]  (already transposed)

    const int tid  = threadIdx.x;
    const int lane = tid & 31;
    const int wave = tid >> 5;
    const int m0 = blockIdx.y * 128;
    const int n0 = blockIdx.x * 128;
    const int wm = (wave & 3) * 32;      // 4 waves down
    const int wn = (wave >> 2) * 64;     // 2 waves across
    const int lr = lane & 15;
    const int hi = lane >> 4;

    // stage one 128x64 A tile + 128x64 Bt tile: 8 async b128 per thread
    auto stage = [&](int buf, int k0) {
#pragma unroll
        for (int t = 0; t < 4; ++t) {
            const int i   = tid + t * 256;
            const int row = i >> 3;
            const int seg = (i & 7) * 8;           // bf16 elements (16B chunks)
            copy16_async(&As[buf][row * 64 + seg],
                         A + (size_t)(m0 + row) * K + k0 + seg);
            copy16_async(&Bs[buf][row * 64 + seg],
                         Bt + (size_t)(n0 + row) * K + k0 + seg);
        }
    };

    v8f acc[2][4] = {};
    const int nk = K >> 6;
    stage(0, 0);
    for (int ks = 0; ks < nk; ++ks) {
        const int cur = ks & 1;
        if (ks + 1 < nk) {
            stage(cur ^ 1, (ks + 1) << 6);
            ASYNC_WAIT(8);                 // current tile's 8 copies done
        } else {
            ASYNC_WAIT(0);
        }
        __syncthreads();
#pragma unroll
        for (int kk = 0; kk < 64; kk += 32) {
            Frag16 a[2], b[4];
#pragma unroll
            for (int s = 0; s < 2; ++s)
#pragma unroll
                for (int i = 0; i < 8; ++i) {
                    const int ka = kk + ((i & 4) ? 16 : 0) + hi * 8 + (i & 3) * 2;
                    a[s].u[i] =
                        *(const unsigned*)&As[cur][(wm + s * 16 + lr) * 64 + ka];
                }
#pragma unroll
            for (int t = 0; t < 4; ++t)
#pragma unroll
                for (int i = 0; i < 8; ++i) {
                    const int kb = kk + ((i & 4) ? 16 : 0) + hi * 8 + (i & 3) * 2;
                    b[t].u[i] =
                        *(const unsigned*)&Bs[cur][(wn + t * 16 + lr) * 64 + kb];
                }
#pragma unroll
            for (int s = 0; s < 2; ++s)
#pragma unroll
                for (int t = 0; t < 4; ++t)
                    acc[s][t] = __builtin_amdgcn_wmma_f32_16x16x32_bf16(
                        false, a[s].v, false, b[t].v, (short)0, acc[s][t],
                        false, false);
        }
        __syncthreads();   // tile fully consumed before it is restaged
    }

    // epilogue: C/D layout -> lane(0-15): M=r, lane(16-31): M=r+8
#pragma unroll
    for (int s = 0; s < 2; ++s) {
#pragma unroll
        for (int t = 0; t < 4; ++t) {
#pragma unroll
            for (int r = 0; r < 8; ++r) {
                const int row = m0 + wm + s * 16 + hi * 8 + r;
                const int col = n0 + wn + t * 16 + lr;
                float v = acc[s][t][r] + bias[col];
                if (EPI == 1) v += resid[(size_t)row * N + col];
                if (EPI == 2) v = v > 0.f ? v : 0.f;
                if (EPI == 1) Cf[(size_t)row * N + col] = v;
                else          Cb[(size_t)row * N + col] = f2bf(v);
            }
        }
    }
}

// ---------------------------------------------------------------------------
// Attention for one head: block per batch n. K,V tiles async-staged into
// dynamic LDS (2 * 43*512 bf16 = 88 KB < 320 KB WGP LDS).
// Out points at the head's 512-col slice of the concatenated (M,1024) buffer.
// ---------------------------------------------------------------------------
__global__ void __launch_bounds__(256)
attn_kernel(const bf16_t* __restrict__ Q, const bf16_t* __restrict__ Kv,
            const bf16_t* __restrict__ Vv, bf16_t* __restrict__ Out) {
    const int n   = blockIdx.x;
    const int tid = threadIdx.x;
    extern __shared__ char smem[];
    bf16_t* Ks = (bf16_t*)smem;            // [43][512]
    bf16_t* Vs = Ks + TT * DKK;            // [43][512]
    __shared__ float scs[48];
    __shared__ float prb[48];
    __shared__ float sinv;

    const size_t base = (size_t)n * TT * DKK;
    const int chunks = TT * DKK / 8;       // 16B chunks
    for (int c = tid; c < chunks; c += 256) {
        copy16_async(&Ks[c * 8], Kv + base + (size_t)c * 8);
        copy16_async(&Vs[c * 8], Vv + base + (size_t)c * 8);
    }
    ASYNC_WAIT(0);
    __syncthreads();

    const float scale = rsqrtf((float)DKK);
    const int wave = tid >> 5, lane = tid & 31;

    for (int t = 0; t < TT; ++t) {
        const bf16_t* qrow = Q + base + (size_t)t * DKK;
        for (int s = wave; s < TT; s += 8) {
            float p = 0.f;
            for (int d = lane; d < DKK; d += 32)
                p += bf2f(qrow[d]) * bf2f(Ks[s * DKK + d]);
#pragma unroll
            for (int off = 16; off; off >>= 1)
                p += __shfl_xor(p, off, 32);
            if (lane == 0) scs[s] = p * scale;
        }
        __syncthreads();
        if (tid == 0) {
            float mx = -1e30f;
            for (int s = 0; s < TT; ++s) mx = fmaxf(mx, scs[s]);
            float sm = 0.f;
            for (int s = 0; s < TT; ++s) {
                float e = __expf(scs[s] - mx);
                prb[s] = e;
                sm += e;
            }
            sinv = 1.f / sm;
        }
        __syncthreads();
        for (int d = tid; d < DVV; d += 256) {
            float acc = 0.f;
            for (int s = 0; s < TT; ++s)
                acc += prb[s] * bf2f(Vs[s * DVV + d]);
            Out[((size_t)n * TT + t) * (2 * DVV) + d] = f2bf(acc * sinv);
        }
        __syncthreads();
    }
}

// ---------------------------------------------------------------------------
// LayerNorm over H=1024 per row (biased variance, eps 1e-5). outb optional.
// Safe when outf == in (all reads of x complete before the write loop).
// ---------------------------------------------------------------------------
__global__ void __launch_bounds__(256)
ln_kernel(const float* __restrict__ in, float* __restrict__ outf,
          bf16_t* __restrict__ outb, const float* __restrict__ g,
          const float* __restrict__ b) {
    const int row = blockIdx.x;
    const int tid = threadIdx.x;
    __shared__ float red[256];
    const float* x = in + (size_t)row * HH;

    float s = 0.f;
    for (int i = tid; i < HH; i += 256) s += x[i];
    red[tid] = s;
    __syncthreads();
    for (int off = 128; off; off >>= 1) {
        if (tid < off) red[tid] += red[tid + off];
        __syncthreads();
    }
    const float mean = red[0] * (1.f / HH);
    __syncthreads();

    float v = 0.f;
    for (int i = tid; i < HH; i += 256) {
        float d = x[i] - mean;
        v += d * d;
    }
    red[tid] = v;
    __syncthreads();
    for (int off = 128; off; off >>= 1) {
        if (tid < off) red[tid] += red[tid + off];
        __syncthreads();
    }
    const float inv = rsqrtf(red[0] * (1.f / HH) + 1e-5f);
    __syncthreads();

    for (int i = tid; i < HH; i += 256) {
        float o = (x[i] - mean) * inv * g[i] + b[i];
        outf[(size_t)row * HH + i] = o;
        if (outb) outb[(size_t)row * HH + i] = f2bf(o);
    }
}

// ---------------------------------------------------------------------------
// Final head: out[n] = sigmoid(h2[n*T + 0, :] . fl_w + fl_b)
// ---------------------------------------------------------------------------
__global__ void __launch_bounds__(256)
final_kernel(const float* __restrict__ h2, const float* __restrict__ flw,
             const float* __restrict__ flb, float* __restrict__ out) {
    const int n   = blockIdx.x;
    const int tid = threadIdx.x;
    __shared__ float red[256];
    const float* x = h2 + (size_t)n * TT * HH;   // CLS row (t = 0)
    float s = 0.f;
    for (int i = tid; i < HH; i += 256) s += x[i] * flw[i];
    red[tid] = s;
    __syncthreads();
    for (int off = 128; off; off >>= 1) {
        if (tid < off) red[tid] += red[tid + off];
        __syncthreads();
    }
    if (tid == 0) out[n] = 1.f / (1.f + __expf(-(red[0] + flb[0])));
}

// ---------------------------------------------------------------------------
// Host launcher
// ---------------------------------------------------------------------------
extern "C" void kernel_launch(void* const* d_in, const int* in_sizes, int n_in,
                              void* d_out, int out_size, void* d_ws,
                              size_t ws_size, hipStream_t stream) {
    (void)in_sizes; (void)n_in; (void)out_size; (void)ws_size;

    const int*   inputs = (const int*)  d_in[0];
    const float* emb    = (const float*)d_in[1];
    const float* pos    = (const float*)d_in[2];
    const float* q1_w = (const float*)d_in[3],  *q1_b = (const float*)d_in[4];
    const float* k1_w = (const float*)d_in[5],  *k1_b = (const float*)d_in[6];
    const float* v1_w = (const float*)d_in[7],  *v1_b = (const float*)d_in[8];
    const float* q2_w = (const float*)d_in[9],  *q2_b = (const float*)d_in[10];
    const float* k2_w = (const float*)d_in[11], *k2_b = (const float*)d_in[12];
    const float* v2_w = (const float*)d_in[13], *v2_b = (const float*)d_in[14];
    const float* proj_w = (const float*)d_in[15], *proj_b = (const float*)d_in[16];
    const float* ln_g = (const float*)d_in[17], *ln_b = (const float*)d_in[18];
    const float* w1_w = (const float*)d_in[19], *w1_b = (const float*)d_in[20];
    const float* w2_w = (const float*)d_in[21], *w2_b = (const float*)d_in[22];
    const float* fl_w = (const float*)d_in[23], *fl_b = (const float*)d_in[24];
    float* out = (float*)d_out;

    // ---- workspace layout
    char* ws = (char*)d_ws;
    size_t off = 0;
    auto take = [&](size_t bytes) {
        size_t o = off;
        off = (off + bytes + 255) & ~(size_t)255;
        return o;
    };
    const size_t szRowF = (size_t)MROW * HH * 4;     // f32 (M,1024)
    const size_t szRowB = (size_t)MROW * HH * 2;     // bf16 (M,1024)
    const size_t szHdB  = (size_t)MROW * DKK * 2;    // bf16 (M,512)

    float*  xf   = (float*) (ws + take(szRowF));
    bf16_t* xb   = (bf16_t*)(ws + take(szRowB));
    size_t  o_qkv = take(6 * szHdB + szRowB);        // QKV + attn region
    bf16_t* q1   = (bf16_t*)(ws + o_qkv);
    bf16_t* k1   = q1 + (size_t)MROW * DKK;
    bf16_t* v1   = k1 + (size_t)MROW * DKK;
    bf16_t* q2   = v1 + (size_t)MROW * DKK;
    bf16_t* k2   = q2 + (size_t)MROW * DKK;
    bf16_t* v2   = k2 + (size_t)MROW * DKK;
    bf16_t* attn = v2 + (size_t)MROW * DKK;          // (M,1024) concat
    bf16_t* fb   = q1;                               // FFN mid aliases dead QKV+attn
    float*  y    = (float*) (ws + take(szRowF));     // proj out / FFN2 out
    float*  hf   = (float*) (ws + take(szRowF));
    bf16_t* hb   = (bf16_t*)(ws + take(szRowB));
    bf16_t* q1wb = (bf16_t*)(ws + take((size_t)HH * DKK * 2));
    bf16_t* k1wb = (bf16_t*)(ws + take((size_t)HH * DKK * 2));
    bf16_t* v1wb = (bf16_t*)(ws + take((size_t)HH * DVV * 2));
    bf16_t* q2wb = (bf16_t*)(ws + take((size_t)HH * DKK * 2));
    bf16_t* k2wb = (bf16_t*)(ws + take((size_t)HH * DKK * 2));
    bf16_t* v2wb = (bf16_t*)(ws + take((size_t)HH * DVV * 2));
    bf16_t* pjwb = (bf16_t*)(ws + take((size_t)2 * DVV * HH * 2));
    bf16_t* w1wb = (bf16_t*)(ws + take((size_t)HH * FFF * 2));
    bf16_t* w2wb = (bf16_t*)(ws + take((size_t)FFF * HH * 2));

    // ---- 1) weight convert + transpose: (K,N) f32 -> (N,K) bf16
    auto cvtT = [&](const float* s, bf16_t* d, int K, int N) {
        cvt_transpose_kernel<<<2048, 256, 0, stream>>>(s, d, K, N);
    };
    cvtT(q1_w, q1wb, HH, DKK);  cvtT(k1_w, k1wb, HH, DKK);
    cvtT(v1_w, v1wb, HH, DVV);  cvtT(q2_w, q2wb, HH, DKK);
    cvtT(k2_w, k2wb, HH, DKK);  cvtT(v2_w, v2wb, HH, DVV);
    cvtT(proj_w, pjwb, 2 * DVV, HH);
    cvtT(w1_w, w1wb, HH, FFF);  cvtT(w2_w, w2wb, FFF, HH);

    // ---- 2) embed
    embed_kernel<<<MROW, 256, 0, stream>>>(inputs, emb, pos, xf, xb);

    const dim3 blk(256);
    const int mt = MROW / 128;                       // 172

    // ---- 3) QKV projections (bias -> bf16)
    {
        const dim3 grd(DKK / 128, mt);
        gemm_wmma_kernel<0><<<grd, blk, 0, stream>>>(xb, q1wb, q1_b, nullptr, nullptr, q1, MROW, DKK, HH);
        gemm_wmma_kernel<0><<<grd, blk, 0, stream>>>(xb, k1wb, k1_b, nullptr, nullptr, k1, MROW, DKK, HH);
        gemm_wmma_kernel<0><<<grd, blk, 0, stream>>>(xb, v1wb, v1_b, nullptr, nullptr, v1, MROW, DVV, HH);
        gemm_wmma_kernel<0><<<grd, blk, 0, stream>>>(xb, q2wb, q2_b, nullptr, nullptr, q2, MROW, DKK, HH);
        gemm_wmma_kernel<0><<<grd, blk, 0, stream>>>(xb, k2wb, k2_b, nullptr, nullptr, k2, MROW, DKK, HH);
        gemm_wmma_kernel<0><<<grd, blk, 0, stream>>>(xb, v2wb, v2_b, nullptr, nullptr, v2, MROW, DVV, HH);
    }

    // ---- 4) attention per head, writing into concat slices of attn
    const size_t attn_smem = (size_t)2 * TT * DKK * sizeof(bf16_t);   // 88 KB
    attn_kernel<<<NB, 256, attn_smem, stream>>>(q1, k1, v1, attn);
    attn_kernel<<<NB, 256, attn_smem, stream>>>(q2, k2, v2, attn + DVV);

    // ---- 5) proj + residual(x) -> y (f32)
    gemm_wmma_kernel<1><<<dim3(HH / 128, mt), blk, 0, stream>>>(
        attn, pjwb, proj_b, xf, y, nullptr, MROW, HH, 2 * DVV);

    // ---- 6) LN1: y -> hf (f32) + hb (bf16)
    ln_kernel<<<MROW, 256, 0, stream>>>(y, hf, hb, ln_g, ln_b);

    // ---- 7) FFN up + ReLU -> fb (bf16)
    gemm_wmma_kernel<2><<<dim3(FFF / 128, mt), blk, 0, stream>>>(
        hb, w1wb, w1_b, nullptr, nullptr, fb, MROW, FFF, HH);

    // ---- 8) FFN down + residual(hf) -> y (f32)
    gemm_wmma_kernel<1><<<dim3(HH / 128, mt), blk, 0, stream>>>(
        fb, w2wb, w2_b, hf, y, nullptr, MROW, HH, FFF);

    // ---- 9) LN2 in-place on y (same gamma/beta per reference)
    ln_kernel<<<MROW, 256, 0, stream>>>(y, y, nullptr, ln_g, ln_b);

    // ---- 10) sigmoid head on CLS rows
    final_kernel<<<NB, 256, 0, stream>>>(y, fl_w, fl_b, out);
}
